// MOEModel_Light_54271206752779
// MI455X (gfx1250) — compile-verified, matrix operand order checked by reading, not run on hardware
//
#include <hip/hip_runtime.h>
#include <hip/hip_bf16.h>
#include <math.h>

// ---------------------------------------------------------------------------
// MI455X (gfx1250). All contractions on V_WMMA_F32_16X16X32_BF16 (wave32,
// fp32 accumulate). ~37 GFLOP + 32MB output => trivially fast on 23.3TB/s HBM
// as long as loads are vectorized (b128) and guards are branchless clamps.
// ---------------------------------------------------------------------------

typedef __attribute__((ext_vector_type(16))) __bf16 v16bf;
typedef __attribute__((ext_vector_type(8)))  __bf16 v8bf;
typedef __attribute__((ext_vector_type(8)))  float  v8f;
typedef __attribute__((ext_vector_type(4)))  float  v4f;

constexpr int B_ = 32, P_ = 500, N_ = 500, D_ = 128, H_ = 8, DK_ = 16;
constexpr int F_ = 512, E_ = 4;
constexpr int T_ = B_ * P_;              // 16000 (exact multiple of 16)
constexpr float INV_SQRT_D  = 0.08838834764831845f;
constexpr float INV_SQRT_DK = 0.25f;

// 16-bit A-operand (16x32) K map (ISA 7.12.2): elems 0-7 -> K = e + 8*hi,
// elems 8-15 -> K = e + 8 + 8*hi. Both 8-element groups are CONTIGUOUS in K.
__device__ __forceinline__ v16bf cat8(v8bf lo, v8bf hi8) {
    return __builtin_shufflevector(lo, hi8, 0,1,2,3,4,5,6,7,8,9,10,11,12,13,14,15);
}
__device__ __forceinline__ v8bf cvt8(v4f a, v4f b) {
    v8bf r;
    r[0]=(__bf16)a[0]; r[1]=(__bf16)a[1]; r[2]=(__bf16)a[2]; r[3]=(__bf16)a[3];
    r[4]=(__bf16)b[0]; r[5]=(__bf16)b[1]; r[6]=(__bf16)b[2]; r[7]=(__bf16)b[3];
    return r;
}
// 8 valid bf16 + 8 zeros (for DK=16 padded into K=32 WMMA operands)
__device__ __forceinline__ v16bf lo8_pad(v4f a, v4f b) {
    v8bf z = {};
    return cat8(cvt8(a, b), z);
}

// ---------------------------------------------------------------------------
// Xcat[t,0:128]=eln, [128:132]=attr, [132:192]=0  (zero-padded concat input)
// ---------------------------------------------------------------------------
__global__ void k_pack(const float* __restrict__ eln, const float* __restrict__ attr,
                       float* __restrict__ Xc) {
    int t = blockIdx.x, k = threadIdx.x;           // 192 threads
    float v = 0.f;
    if (k < 128)      v = eln[(size_t)t * 128 + k];
    else if (k < 132) v = attr[(size_t)t * 4 + (k - 128)];
    Xc[(size_t)t * 192 + k] = v;
}

// ---------------------------------------------------------------------------
// Y = X@W (+bias)(+addsrc)(relu) or gates: Y += g[row,eidx]*(XW+bias).
// X:[Trows,Din], W:[Kw<=Din rows, Dout]. Trows%16==0, Dout%128==0 => no edge
// guards. K staged in 64-chunks; W transposed in LDS; WMMA operands read as
// two 16B LDS vectors each. grid=(Trows/16, Dout/128), block=256 (8 waves).
// ---------------------------------------------------------------------------
__global__ void k_gemm(const float* __restrict__ X, const float* __restrict__ W,
                       const float* __restrict__ bias, const float* __restrict__ addsrc,
                       const float* __restrict__ gates, int eidx,
                       float* __restrict__ Y, int Din, int Kw, int Dout, int relu) {
    __shared__ __bf16 xs[16][72];        // padded stride: bank spread, 16B aligned
    __shared__ __bf16 wshT[128][72];     // W chunk, transposed [n][k]
    int tid = threadIdx.x, lane = tid & 31, wv = tid >> 5;
    int hi = lane >> 4, ln = lane & 15;
    int m0 = blockIdx.x * 16, n0 = blockIdx.y * 128;
    v8f acc = {};
    for (int k0 = 0; k0 < Din; k0 += 64) {
        if (k0 + 64 < Din)
            __builtin_prefetch(&W[(size_t)(k0 + 64) * Dout + n0], 0, 3);
        {   // X tile 16x64: one float4 per thread (1024 elems = 256*4)
            int basei = tid * 4;
            int m = basei >> 6, k = basei & 63;
            v4f xv = *(const v4f*)&X[(size_t)(m0 + m) * Din + k0 + k];
            xs[m][k]   = (__bf16)xv[0];
            xs[m][k+1] = (__bf16)xv[1];
            xs[m][k+2] = (__bf16)xv[2];
            xs[m][k+3] = (__bf16)xv[3];
        }
#pragma unroll
        for (int it = 0; it < 8; ++it) {   // W chunk 64x128: 8 float4 per thread
            int basei = it * 1024 + tid * 4;
            int k = basei >> 7, n = basei & 127;
            int krow = k0 + k; krow = krow < Kw ? krow : Kw - 1;  // pad rows: X is 0 there
            v4f wvv = *(const v4f*)&W[(size_t)krow * Dout + n0 + n];
            wshT[n][k]   = (__bf16)wvv[0];
            wshT[n+1][k] = (__bf16)wvv[1];
            wshT[n+2][k] = (__bf16)wvv[2];
            wshT[n+3][k] = (__bf16)wvv[3];
        }
        __syncthreads();
        int col = wv * 16 + ln;
#pragma unroll
        for (int kk = 0; kk < 64; kk += 32) {
            v16bf a = cat8(*(const v8bf*)&xs[ln][kk + 8*hi],
                           *(const v8bf*)&xs[ln][kk + 16 + 8*hi]);
            v16bf bm = cat8(*(const v8bf*)&wshT[col][kk + 8*hi],
                            *(const v8bf*)&wshT[col][kk + 16 + 8*hi]);
            acc = __builtin_amdgcn_wmma_f32_16x16x32_bf16(false, a, false, bm,
                                                          (short)0, acc, false, false);
        }
        __syncthreads();
    }
    int n = n0 + wv * 16 + ln;
#pragma unroll
    for (int j = 0; j < 8; ++j) {
        size_t idx = (size_t)(m0 + j + 8 * hi) * Dout + n;   // C: m=j+8*hi, n=lane&15
        float v = acc[j];
        if (bias) v += bias[n];
        if (gates) {
            Y[idx] += gates[(size_t)(m0 + j + 8 * hi) * E_ + eidx] * v;
        } else {
            if (addsrc) v += addsrc[idx];
            if (relu && v < 0.f) v = 0.f;
            Y[idx] = v;
        }
    }
}

// ---------------------------------------------------------------------------
// Flash attention, one wave per (p-tile, head, batch). DK=16 zero-padded to
// K=32: each lane's valid elements are 8 contiguous floats -> 2x b128 loads.
// Clamped loads are safe: garbage rows are store-guarded, garbage score
// columns are forced to -1e30, and P columns for n>=N are exactly 0 in P@V.
// ---------------------------------------------------------------------------
__global__ void k_attn(const float* __restrict__ Qb, const float* __restrict__ Kb,
                       const float* __restrict__ Vb, const float* __restrict__ mask,
                       float* __restrict__ att) {
    __shared__ __bf16 pls[16][24];       // padded stride (48B): conflict-free
    int pt = blockIdx.x, h = blockIdx.y, b = blockIdx.z;
    int lane = threadIdx.x;
    int hi = lane >> 4, ln = lane & 15;
    int p0 = pt * 16;
    int prowA = p0 + ln, prowC = prowA < P_ ? prowA : P_ - 1;
    const float* qrow = Qb + ((size_t)(b * P_ + prowC)) * D_ + h * DK_ + hi * 8;
    v16bf aq = lo8_pad(*(const v4f*)qrow, *(const v4f*)(qrow + 4));
    v8f czero = {};
    float mrow[8], lrow[8];
    v8f acc = {};
#pragma unroll
    for (int j = 0; j < 8; ++j) { mrow[j] = -1e30f; lrow[j] = 0.f; }

    for (int nt = 0; nt < 32; ++nt) {
        int nB = nt * 16 + ln, nC = nB < N_ ? nB : N_ - 1;
        const float* krow = Kb + ((size_t)(b * N_ + nC)) * D_ + h * DK_ + hi * 8;
        v16bf bk = lo8_pad(*(const v4f*)krow, *(const v4f*)(krow + 4));
        v8f s = __builtin_amdgcn_wmma_f32_16x16x32_bf16(false, aq, false, bk,
                                                        (short)0, czero, false, false);
        float pv[8];
#pragma unroll
        for (int j = 0; j < 8; ++j) {
            int prow = p0 + j + 8 * hi;
            int pc = prow < P_ ? prow : P_ - 1;
            float mval = mask[((size_t)(b * P_ + pc)) * N_ + nC];
            float sc = (nB < N_ && prow < P_) ? s[j] * INV_SQRT_DK + mval : -1e30f;
            float rm = sc;
            rm = fmaxf(rm, __shfl_xor(rm, 1, 32));
            rm = fmaxf(rm, __shfl_xor(rm, 2, 32));
            rm = fmaxf(rm, __shfl_xor(rm, 4, 32));
            rm = fmaxf(rm, __shfl_xor(rm, 8, 32));
            float mnew = fmaxf(mrow[j], rm);
            float corr = __expf(mrow[j] - mnew);
            float pij  = __expf(sc - mnew);
            float rs = pij;
            rs += __shfl_xor(rs, 1, 32);
            rs += __shfl_xor(rs, 2, 32);
            rs += __shfl_xor(rs, 4, 32);
            rs += __shfl_xor(rs, 8, 32);
            lrow[j] = lrow[j] * corr + rs;
            mrow[j] = mnew;
            acc[j] *= corr;
            pv[j] = pij;
        }
        __syncthreads();
#pragma unroll
        for (int j = 0; j < 8; ++j) pls[j + 8 * hi][ln] = (__bf16)pv[j];
        __syncthreads();
        v8bf z = {};
        v16bf ap = cat8(*(const v8bf*)&pls[ln][8 * hi], z);
        v16bf bv = {};
#pragma unroll
        for (int e = 0; e < 8; ++e) {          // V rows strided; masked cols have P==0
            int kk = e + 8 * hi;
            int nr = nt * 16 + kk; nr = nr < N_ ? nr : N_ - 1;
            bv[e] = (__bf16)Vb[((size_t)(b * N_ + nr)) * D_ + h * DK_ + ln];
        }
        acc = __builtin_amdgcn_wmma_f32_16x16x32_bf16(false, ap, false, bv,
                                                      (short)0, acc, false, false);
    }
#pragma unroll
    for (int j = 0; j < 8; ++j) {
        int prow = p0 + j + 8 * hi;
        if (prow < P_)
            att[((size_t)(b * P_ + prow)) * D_ + h * DK_ + ln] = acc[j] / lrow[j];
    }
}

// InstanceNorm over P per (b,d); lane=d -> coalesced.
__global__ void k_inorm(float* __restrict__ x, const float* __restrict__ s,
                        const float* __restrict__ bsh) {
    int d = threadIdx.x, b = blockIdx.x;
    const size_t base = (size_t)b * P_ * D_ + d;
    float sum = 0.f, sq = 0.f;
    for (int p = 0; p < P_; ++p) {
        float v = x[base + (size_t)p * D_];
        sum += v; sq += v * v;
    }
    float mean = sum / (float)P_;
    float var  = sq / (float)P_ - mean * mean;
    float rs = rsqrtf(var + 1e-5f);
    float sc = s[d], bb = bsh[d];
    for (int p = 0; p < P_; ++p) {
        size_t i = base + (size_t)p * D_;
        x[i] = (x[i] - mean) * rs * sc + bb;
    }
}

__global__ void k_pool_gate(const float* __restrict__ q, const float* __restrict__ gW,
                            float* __restrict__ gbuf, float* __restrict__ imp,
                            float* __restrict__ lossb) {
    __shared__ float pooled[128];
    int d = threadIdx.x;
    float s = 0.f;
    for (int t = 0; t < T_; ++t) s += q[(size_t)t * D_ + d];
    pooled[d] = s / (float)T_;
    __syncthreads();
    if (d == 0) {
        float l0 = 0.f, l1 = 0.f;
        for (int i = 0; i < 128; ++i) { l0 += pooled[i] * gW[i*2]; l1 += pooled[i] * gW[i*2+1]; }
        float m = fmaxf(l0, l1);
        float e0 = __expf(l0 - m), e1 = __expf(l1 - m), z = e0 + e1;
        gbuf[0] = e0 / z; gbuf[1] = e1 / z; gbuf[2] = (l1 > l0) ? 1.f : 0.f;
        imp[0] = imp[1] = imp[2] = imp[3] = 0.f;
        lossb[0] = 0.f;
    }
}

__global__ void k_moe_gates(const float* __restrict__ x, const float* __restrict__ wg,
                            float* __restrict__ gates, float* __restrict__ imp) {
    int t = blockIdx.x * 128 + threadIdx.x;
    if (t >= T_) return;
    float lg[4] = {0.f, 0.f, 0.f, 0.f};
    for (int i = 0; i < 128; ++i) {
        float v = x[(size_t)t * D_ + i];
        lg[0] += v * wg[i*4+0]; lg[1] += v * wg[i*4+1];
        lg[2] += v * wg[i*4+2]; lg[3] += v * wg[i*4+3];
    }
    int i0 = 0;
    for (int e = 1; e < 4; ++e) if (lg[e] > lg[i0]) i0 = e;
    int i1 = -1;
    for (int e = 0; e < 4; ++e) { if (e == i0) continue; if (i1 < 0 || lg[e] > lg[i1]) i1 = e; }
    float m = fmaxf(lg[i0], lg[i1]);
    float e0 = __expf(lg[i0] - m), e1 = __expf(lg[i1] - m), z = e0 + e1;
    for (int e = 0; e < 4; ++e) gates[(size_t)t * E_ + e] = 0.f;
    gates[(size_t)t * E_ + i0] = e0 / z;
    gates[(size_t)t * E_ + i1] = e1 / z;
    atomicAdd(&imp[i0], e0 / z);
    atomicAdd(&imp[i1], e1 / z);
}

__global__ void k_moe_loss(const float* __restrict__ imp, const float* __restrict__ gbuf,
                           float* __restrict__ out_last, float* __restrict__ lossb) {
    if (threadIdx.x == 0 && blockIdx.x == 0) {
        float m = (imp[0] + imp[1] + imp[2] + imp[3]) * 0.25f;
        float v = 0.f;
        for (int e = 0; e < 4; ++e) { float d = imp[e] - m; v += d * d; }
        v *= 0.25f;
        float loss = v / (m * m + 1e-10f);
        lossb[0] = loss;
        out_last[0] = (gbuf[2] > 0.5f) ? loss : 0.f;
    }
}

__global__ void k_zero(float* __restrict__ p, size_t n) {
    size_t i = (size_t)blockIdx.x * blockDim.x + threadIdx.x;
    if (i < n) p[i] = 0.f;
}

__global__ void k_select_scale(const float* __restrict__ dense, const float* __restrict__ moe,
                               const float* __restrict__ gbuf, float* __restrict__ mh, int n) {
    int i = blockIdx.x * 256 + threadIdx.x;
    if (i >= n) return;
    int sel = (gbuf[2] > 0.5f) ? 1 : 0;
    mh[i] = (sel ? moe[i] : dense[i]) * gbuf[sel];
}

// ---------------------------------------------------------------------------
// Pointer scores + tanh clip + mask + row softmax. A tile LDS-staged bf16,
// shared by 8 waves; both WMMA operands read as 16B vectors.
// ---------------------------------------------------------------------------
__global__ void k_pointer(const float* __restrict__ mh, const float* __restrict__ enc,
                          const float* __restrict__ mask, float* __restrict__ out) {
    __shared__ float  sc[16][512];       // 32 KB
    __shared__ __bf16 amat[16][136];     // padded stride (272B), 16B aligned
    int b = blockIdx.y, pt = blockIdx.x, p0 = pt * 16;
    int tid = threadIdx.x, lane = tid & 31, wv = tid >> 5;
    int hi = lane >> 4, ln = lane & 15;
    {   // stage A (mh tile): 8 consecutive floats per thread (2048 = 256*8)
        int basei = tid * 8;
        int m = basei >> 7, k = basei & 127;
        int row = p0 + m; row = row < P_ ? row : P_ - 1;      // garbage rows guarded later
        const float* src = &mh[((size_t)(b * P_ + row)) * D_ + k];
        v4f x0 = *(const v4f*)src, x1 = *(const v4f*)(src + 4);
        amat[m][k]   = (__bf16)x0[0]; amat[m][k+1] = (__bf16)x0[1];
        amat[m][k+2] = (__bf16)x0[2]; amat[m][k+3] = (__bf16)x0[3];
        amat[m][k+4] = (__bf16)x1[0]; amat[m][k+5] = (__bf16)x1[1];
        amat[m][k+6] = (__bf16)x1[2]; amat[m][k+7] = (__bf16)x1[3];
    }
    __syncthreads();
    for (int nt = wv; nt < 32; nt += 8) {
        int n = nt * 16 + ln, nC = n < N_ ? n : N_ - 1;
        const float* erow = enc + ((size_t)(b * N_ + nC)) * D_;
        v8f acc = {};
#pragma unroll
        for (int kb = 0; kb < 128; kb += 32) {
            v16bf a = cat8(*(const v8bf*)&amat[ln][kb + 8*hi],
                           *(const v8bf*)&amat[ln][kb + 16 + 8*hi]);
            const float* e0 = erow + kb + 8 * hi;
            v16bf bm = cat8(cvt8(*(const v4f*)e0,        *(const v4f*)(e0 + 4)),
                            cvt8(*(const v4f*)(e0 + 16), *(const v4f*)(e0 + 20)));
            acc = __builtin_amdgcn_wmma_f32_16x16x32_bf16(false, a, false, bm,
                                                          (short)0, acc, false, false);
        }
#pragma unroll
        for (int j = 0; j < 8; ++j) {
            int m = j + 8 * hi, p = p0 + m;
            int pc = p < P_ ? p : P_ - 1;
            float mval = mask[((size_t)(b * P_ + pc)) * N_ + nC];
            float s = (n < N_ && p < P_)
                      ? 10.f * tanhf(acc[j] * INV_SQRT_D) + mval : -1e30f;
            sc[m][nt * 16 + ln] = s;
        }
    }
    __syncthreads();
    int r = tid >> 4, c = tid & 15;
    float mx = -1e30f;
    for (int n = c; n < 512; n += 16) mx = fmaxf(mx, sc[r][n]);
    for (int off = 1; off < 16; off <<= 1) mx = fmaxf(mx, __shfl_xor(mx, off, 32));
    float sum = 0.f;
    for (int n = c; n < 512; n += 16) { float e = __expf(sc[r][n] - mx); sc[r][n] = e; sum += e; }
    for (int off = 1; off < 16; off <<= 1) sum += __shfl_xor(sum, off, 32);
    float inv = 1.f / sum;
    int p = p0 + r;
    if (p < P_)
        for (int n = c; n < N_; n += 16)
            out[((size_t)(b * P_ + p)) * N_ + n] = sc[r][n] * inv;
}

// ---------------------------------------------------------------------------
extern "C" void kernel_launch(void* const* d_in, const int* in_sizes, int n_in,
                              void* d_out, int out_size, void* d_ws, size_t ws_size,
                              hipStream_t stream) {
    (void)in_sizes; (void)n_in; (void)out_size; (void)ws_size;
    const float* eln    = (const float*)d_in[0];
    const float* attr   = (const float*)d_in[1];
    const float* mask   = (const float*)d_in[2];
    const float* enc    = (const float*)d_in[3];
    const float* WqLast = (const float*)d_in[4];
    const float* lWq    = (const float*)d_in[5];
    const float* lWk    = (const float*)d_in[6];
    const float* lWv    = (const float*)d_in[7];
    const float* lWcW   = (const float*)d_in[8];
    const float* lWcB   = (const float*)d_in[9];
    const float* ln1s   = (const float*)d_in[10];
    const float* ln1b   = (const float*)d_in[11];
    const float* lW1    = (const float*)d_in[12];
    const float* lb1    = (const float*)d_in[13];
    const float* lW2    = (const float*)d_in[14];
    const float* lb2    = (const float*)d_in[15];
    const float* ln2s   = (const float*)d_in[16];
    const float* ln2b   = (const float*)d_in[17];
    const float* gateW  = (const float*)d_in[18];
    const float* wgate  = (const float*)d_in[19];
    const float* moeWe  = (const float*)d_in[20];
    const float* moeBe  = (const float*)d_in[21];
    const float* denseW = (const float*)d_in[22];
    const float* denseB = (const float*)d_in[23];
    float* out = (float*)d_out;

    float* p = (float*)d_ws;
    float* q    = p; p += (size_t)T_ * D_;
    float* Qb   = p; p += (size_t)T_ * D_;
    float* Kb   = p; p += (size_t)T_ * D_;
    float* Vb   = p; p += (size_t)T_ * D_;
    float* att  = p; p += (size_t)T_ * D_;
    float* t1   = p; p += (size_t)T_ * D_;
    float* t2   = p; p += (size_t)T_ * F_;   // also reused as Xcat[T,192] pre-loop
    float* dns  = p; p += (size_t)T_ * D_;
    float* moe  = p; p += (size_t)T_ * D_;
    float* mh   = p; p += (size_t)T_ * D_;
    float* gts  = p; p += (size_t)T_ * E_;
    float* gbuf = p; p += 4;
    float* imp  = p; p += 4;
    float* lsb  = p; p += 4;

    const int MT = T_ / 16;   // 1000 row tiles (exact)

    // q = [eln;attr] @ Wq_last, via zero-padded pack (Din=192, Kw=132)
    k_pack<<<dim3(T_), 192, 0, stream>>>(eln, attr, t2);
    k_gemm<<<dim3(MT, 1), 256, 0, stream>>>(t2, WqLast, nullptr, nullptr, nullptr, 0, q, 192, 132, 128, 0);

    for (int i = 0; i < 3; ++i) {
        k_gemm<<<dim3(MT, 1), 256, 0, stream>>>(q,   lWq + (size_t)i*D_*D_, nullptr, nullptr, nullptr, 0, Qb, D_, D_, D_, 0);
        k_gemm<<<dim3(MT, 1), 256, 0, stream>>>(enc, lWk + (size_t)i*D_*D_, nullptr, nullptr, nullptr, 0, Kb, D_, D_, D_, 0);
        k_gemm<<<dim3(MT, 1), 256, 0, stream>>>(enc, lWv + (size_t)i*D_*D_, nullptr, nullptr, nullptr, 0, Vb, D_, D_, D_, 0);
        k_attn<<<dim3(32, H_, B_), 32, 0, stream>>>(Qb, Kb, Vb, mask, att);
        k_gemm<<<dim3(MT, 1), 256, 0, stream>>>(att, lWcW + (size_t)i*D_*D_, lWcB + i*D_, q, nullptr, 0, t1, D_, D_, D_, 0);
        k_inorm<<<B_, D_, 0, stream>>>(t1, ln1s + i*D_, ln1b + i*D_);
        k_gemm<<<dim3(MT, 4), 256, 0, stream>>>(t1, lW1 + (size_t)i*D_*F_, lb1 + i*F_, nullptr, nullptr, 0, t2, D_, D_, F_, 1);
        k_gemm<<<dim3(MT, 1), 256, 0, stream>>>(t2, lW2 + (size_t)i*F_*D_, lb2 + i*D_, t1, nullptr, 0, q, F_, F_, D_, 0);
        k_inorm<<<B_, D_, 0, stream>>>(q, ln2s + i*D_, ln2b + i*D_);
    }
    k_pool_gate<<<1, 128, 0, stream>>>(q, gateW, gbuf, imp, lsb);
    k_moe_gates<<<(T_ + 127) / 128, 128, 0, stream>>>(q, wgate, gts, imp);
    k_zero<<<(int)(((size_t)T_ * D_ + 255) / 256), 256, 0, stream>>>(moe, (size_t)T_ * D_);
    for (int e = 0; e < 4; ++e)
        k_gemm<<<dim3(MT, 1), 256, 0, stream>>>(q, moeWe + (size_t)e*D_*D_, moeBe + e*D_, nullptr, gts, e, moe, D_, D_, D_, 0);
    k_gemm<<<dim3(MT, 1), 256, 0, stream>>>(q, denseW, denseB, nullptr, nullptr, 0, dns, D_, D_, D_, 0);
    k_moe_loss<<<1, 1, 0, stream>>>(imp, gbuf, out + (size_t)B_ * P_ * N_, lsb);
    k_select_scale<<<(T_ * D_ + 255) / 256, 256, 0, stream>>>(dns, moe, gbuf, mh, T_ * D_);
    k_pointer<<<dim3(32, B_), 256, 0, stream>>>(mh, enc, mask, out);
}
